// NTKSVGP_85624468013105
// MI455X (gfx1250) — matrix operand order repaired and checked by reading, not compiled, not run on hardware
//
#include <hip/hip_runtime.h>

// ---------------------------------------------------------------------------
// Problem dimensions (match reference)
// ---------------------------------------------------------------------------
static constexpr int Nn  = 1024;
static constexpr int Dd  = 256;
static constexpr int Hh  = 384;
static constexpr int Cc  = 10;
static constexpr int Mm  = 128;
static constexpr int NTt = 256;

static constexpr float S2F     = 1.0f;
static constexpr float DELTAF  = 1.0f;
static constexpr float JITTERF = 1e-3f;
static constexpr float L2D     = 2.0f / S2F;
static constexpr float SCALEF  = 1.0f / (DELTAF * (float)Nn);

typedef __attribute__((ext_vector_type(16))) _Float16 v16h;
typedef __attribute__((ext_vector_type(8)))  _Float16 v8h;
typedef __attribute__((ext_vector_type(8)))  float    v8f;

// ---------------------------------------------------------------------------
// WMMA GEMM: C[m,n] = sum_k A[m,k] * B[n,k]   (A: Md x Kd, B: Nd x Kd, both
// row-major f16; C row-major f32). One wave -> 16x32 output strip, two
// v_wmma_f32_16x16x32_f16 per K-step sharing the A fragment.
// Requires Md%16==0, Nd%32==0, Kd%32==0 (true for every call site).
//
// Fragment layouts used here (contiguous K order, same for A and B; not the ISA table order):
//   A (16x32 f16): lane L holds row L%16; lanes 0-15 -> K 0..15,
//                  lanes 16-31 -> K 16..31.
//   B (32x16 f16): lane L holds column L%16; lanes 0-15 -> K 0..15,
//                  lanes 16-31 -> K 16..31. For A*B^T, column n of the B
//                  matrix is row n of operand B -> contiguous row loads.
//   C/D (16x16 f32): lane L holds column L%16, VGPR v -> row v + 8*(L>=16).
// ---------------------------------------------------------------------------
__global__ void __launch_bounds__(128)
wmma_gemm_abt(const _Float16* __restrict__ A, const _Float16* __restrict__ B,
              float* __restrict__ C, int Md, int Nd, int Kd,
              int lda, int ldb, int ldc)
{
    const int wavesPerBlock = blockDim.x >> 5;
    int wid = blockIdx.x * wavesPerBlock + (threadIdx.x >> 5);
    int Nt2 = Nd >> 5;
    int Mt  = Md >> 4;
    if (wid >= Mt * Nt2) return;
    int tm = wid / Nt2;
    int tn = wid % Nt2;

    int lane = threadIdx.x & 31;
    int r    = lane & 15;
    int hi   = lane >> 4;

    const _Float16* Arow = A + (size_t)(tm * 16 + r) * lda;
    const _Float16* B0   = B + (size_t)(tn * 32 + r) * ldb;
    const _Float16* B1   = B + (size_t)(tn * 32 + 16 + r) * ldb;

    v8f acc0 = {}; v8f acc1 = {};
    for (int k0 = 0; k0 < Kd; k0 += 32) {
        v16h a = *(const v16h*)(Arow + k0 + hi * 16);
        v16h b0 = *(const v16h*)(B0 + k0 + hi * 16);
        v16h b1 = *(const v16h*)(B1 + k0 + hi * 16);
        acc0 = __builtin_amdgcn_wmma_f32_16x16x32_f16(false, a, false, b0,
                                                      (short)0, acc0, false, false);
        acc1 = __builtin_amdgcn_wmma_f32_16x16x32_f16(false, a, false, b1,
                                                      (short)0, acc1, false, false);
    }
#pragma unroll
    for (int v = 0; v < 8; ++v) {
        int row = tm * 16 + hi * 8 + v;
        C[(size_t)row * ldc + tn * 32 + r]      = acc0[v];
        C[(size_t)row * ldc + tn * 32 + 16 + r] = acc1[v];
    }
}

// ---------------------------------------------------------------------------
// Elementwise / small kernels
// ---------------------------------------------------------------------------
__global__ void cast_f32_f16(const float* __restrict__ in, _Float16* __restrict__ out, int n)
{
    int i = blockIdx.x * blockDim.x + threadIdx.x;
    if (i < n) out[i] = (_Float16)in[i];
}

// W1 [D,H] (row-major) -> W1T f16 [H,D]
__global__ void transpose_f16(const float* __restrict__ in, _Float16* __restrict__ out,
                              int rows, int cols)  // rows=D, cols=H
{
    int i = blockIdx.x * blockDim.x + threadIdx.x;
    if (i >= rows * cols) return;
    int h = i / rows;
    int d = i % rows;
    out[(size_t)h * rows + d] = (_Float16)in[(size_t)d * cols + h];
}

// h = tanh(a + b1), t = 1 - h^2 (both f16)
__global__ void hidden_act(const float* __restrict__ Apre, const float* __restrict__ b1,
                           _Float16* __restrict__ Hout, _Float16* __restrict__ Tout,
                           int total, int Hd)
{
    int i = blockIdx.x * blockDim.x + threadIdx.x;
    if (i >= total) return;
    float h = tanhf(Apre[i] + b1[i % Hd]);
    Hout[i] = (_Float16)h;
    Tout[i] = (_Float16)(1.0f - h * h);
}

// F = H_X @ W2 + b2 ; lambda1 = -(F - Y)/s2 + F*l2d
__global__ void lambda_kernel(const _Float16* __restrict__ HX, const float* __restrict__ W2,
                              const float* __restrict__ b2, const float* __restrict__ Y,
                              float* __restrict__ lam, int n_rows, int Hd, int Ccols)
{
    int i = blockIdx.x * blockDim.x + threadIdx.x;
    if (i >= n_rows * Ccols) return;
    int n = i / Ccols;
    int c = i % Ccols;
    float f = b2[c];
    const _Float16* hr = HX + (size_t)n * Hd;
    for (int k = 0; k < Hd; ++k) f += (float)hr[k] * W2[(size_t)k * Ccols + c];
    lam[i] = -(f - Y[i]) / S2F + f * L2D;
}

// U = T * W2[:, cls]
__global__ void ubuild(const _Float16* __restrict__ T, const float* __restrict__ W2,
                       int cls, int Ccols, _Float16* __restrict__ U, int total, int Hd)
{
    int i = blockIdx.x * blockDim.x + threadIdx.x;
    if (i >= total) return;
    U[i] = (_Float16)((float)T[i] * W2[(size_t)(i % Hd) * Ccols + cls]);
}

// K = scale*((dot+1)*g + hdot + 1) (+ jitter on diagonal)
__global__ void assemble_k(const float* __restrict__ dots, const float* __restrict__ hdots,
                           const float* __restrict__ gdots, float* __restrict__ K,
                           int total, int cols, float scale, float jitter, int addJitter)
{
    int i = blockIdx.x * blockDim.x + threadIdx.x;
    if (i >= total) return;
    float v = scale * ((dots[i] + 1.0f) * gdots[i] + hdots[i] + 1.0f);
    if (addJitter && (i / cols) == (i % cols)) v += jitter;
    K[i] = v;
}

__global__ void rowsq_f32(const float* __restrict__ X, float* __restrict__ out,
                          int rows, int cols)
{
    int r = blockIdx.x * blockDim.x + threadIdx.x;
    if (r >= rows) return;
    const float* p = X + (size_t)r * cols;
    float s = 0.0f;
    for (int j = 0; j < cols; ++j) s += p[j] * p[j];
    out[r] = s;
}

__global__ void rowsq_f16(const _Float16* __restrict__ X, float* __restrict__ out,
                          int rows, int cols)
{
    int r = blockIdx.x * blockDim.x + threadIdx.x;
    if (r >= rows) return;
    const _Float16* p = X + (size_t)r * cols;
    float s = 0.0f;
    for (int j = 0; j < cols; ++j) { float v = (float)p[j]; s += v * v; }
    out[r] = s;
}

__global__ void ktt_diag(const float* __restrict__ tdot, const float* __restrict__ htdot,
                         const float* __restrict__ gtd, float* __restrict__ Kttd,
                         int n, float scale)
{
    int i = blockIdx.x * blockDim.x + threadIdx.x;
    if (i < n) Kttd[i] = scale * ((tdot[i] + 1.0f) * gtd[i] + htdot[i] + 1.0f);
}

// alpha[m] = sum_n Kzx[m,n] * lambda[n, cls]
__global__ void __launch_bounds__(128)
alpha_kernel(const float* __restrict__ Kzx, const float* __restrict__ lam,
             int cls, int Ccols, float* __restrict__ alpha, int Mrows, int Ncols)
{
    int m = threadIdx.x;
    if (m >= Mrows) return;
    const float* row = Kzx + (size_t)m * Ncols;
    float s = 0.0f;
    for (int n = 0; n < Ncols; ++n) s += row[n] * lam[(size_t)n * Ccols + cls];
    alpha[m] = s;
}

__global__ void combine_B(const float* __restrict__ Kzz, const float* __restrict__ betaraw,
                          float l2d, float* __restrict__ Bm, int n)
{
    int i = blockIdx.x * blockDim.x + threadIdx.x;
    if (i < n) Bm[i] = Kzz[i] + l2d * betaraw[i];
}

__global__ void sub_mat(const float* __restrict__ A, const float* __restrict__ B,
                        float* __restrict__ Cm, int n)
{
    int i = blockIdx.x * blockDim.x + threadIdx.x;
    if (i < n) Cm[i] = A[i] - B[i];
}

// y = A (128x128) @ x
__global__ void __launch_bounds__(128)
matvec128(const float* __restrict__ A, const float* __restrict__ x, float* __restrict__ y)
{
    int r = threadIdx.x;
    const float* row = A + (size_t)r * 128;
    float s = 0.0f;
    for (int j = 0; j < 128; ++j) s += row[j] * x[j];
    y[r] = s;
}

// ---------------------------------------------------------------------------
// 128x128 SPD Cholesky + explicit inverse entirely in LDS (one workgroup,
// 128 threads = 4 waves; 64KB LDS out of the WGP's 320KB).
// ---------------------------------------------------------------------------
__global__ void __launch_bounds__(128)
chol_inverse_128(const float* __restrict__ A, float* __restrict__ Ainv)
{
    __shared__ float sA[128][128];
    int tid = threadIdx.x;
    for (int r = 0; r < 128; ++r) sA[r][tid] = A[(size_t)r * 128 + tid];
    __syncthreads();

    // right-looking Cholesky (lower triangle)
    for (int k = 0; k < 128; ++k) {
        if (tid == 0) sA[k][k] = sqrtf(sA[k][k]);
        __syncthreads();
        if (tid > k) sA[tid][k] /= sA[k][k];
        __syncthreads();
        if (tid > k) {
            float lik = sA[tid][k];
            for (int j = k + 1; j <= tid; ++j) sA[tid][j] -= lik * sA[j][k];
        }
        __syncthreads();
    }

    // thread `tid` computes column tid of A^-1: solve L y = e_tid, L^T x = y
    float y[128];
    for (int r = 0; r < 128; ++r) {
        float s = (r == tid) ? 1.0f : 0.0f;
        for (int j = 0; j < r; ++j) s -= sA[r][j] * y[j];
        y[r] = s / sA[r][r];
    }
    for (int r = 127; r >= 0; --r) {
        float s = y[r];
        for (int j = r + 1; j < 128; ++j) s -= sA[j][r] * y[j];
        y[r] = s / sA[r][r];
    }
    for (int r = 0; r < 128; ++r) Ainv[(size_t)r * 128 + tid] = y[r];
}

// mean/var epilogue: one thread per test point
__global__ void __launch_bounds__(256)
mean_var(const float* __restrict__ Ktz, const float* __restrict__ v,
         const float* __restrict__ tmpm, const float* __restrict__ Kttd,
         float* __restrict__ out, int cls)
{
    int t = blockIdx.x * blockDim.x + threadIdx.x;
    if (t >= NTt) return;
    const float* row = Ktz + (size_t)t * Mm;
    float mean = 0.0f;
    for (int m = 0; m < Mm; ++m) mean += row[m] * v[m];
    float q = 0.0f;
    for (int m = 0; m < Mm; ++m) {
        const float* tr = tmpm + (size_t)m * Mm;
        float p = 0.0f;
        for (int j = 0; j < Mm; ++j) p += tr[j] * row[j];
        q += row[m] * p;
    }
    out[(size_t)t * Cc + cls]                       = mean;         // f_mean
    out[(size_t)NTt * Cc + (size_t)t * Cc + cls]    = Kttd[t] - q;  // f_var
}

// ---------------------------------------------------------------------------
// Host orchestration
// ---------------------------------------------------------------------------
static inline void launch_gemm(const _Float16* A, const _Float16* B, float* C,
                               int Md, int Nd, int Kd, int lda, int ldb, int ldc,
                               hipStream_t s)
{
    int waves  = (Md / 16) * (Nd / 32);
    int blocks = (waves + 3) / 4;
    hipLaunchKernelGGL(wmma_gemm_abt, dim3(blocks), dim3(128), 0, s,
                       A, B, C, Md, Nd, Kd, lda, ldb, ldc);
}

static inline dim3 gsz(int n) { return dim3((n + 255) / 256); }

extern "C" void kernel_launch(void* const* d_in, const int* in_sizes, int n_in,
                              void* d_out, int out_size, void* d_ws, size_t ws_size,
                              hipStream_t stream)
{
    (void)in_sizes; (void)n_in; (void)out_size; (void)ws_size;
    const float* X  = (const float*)d_in[0];
    const float* Y  = (const float*)d_in[1];
    const float* Z  = (const float*)d_in[2];
    const float* Xt = (const float*)d_in[3];
    const float* W1 = (const float*)d_in[4];
    const float* b1 = (const float*)d_in[5];
    const float* W2 = (const float*)d_in[6];
    const float* b2 = (const float*)d_in[7];
    float* out = (float*)d_out;

    // workspace carve-up (256B aligned)
    char* base = (char*)d_ws;
    size_t off = 0;
    auto alloc = [&](size_t bytes) -> void* {
        void* p = base + off;
        off += (bytes + 255) & ~(size_t)255;
        return p;
    };
    _Float16* Xh   = (_Float16*)alloc((size_t)Nn  * Dd * 2);
    _Float16* Zh   = (_Float16*)alloc((size_t)Mm  * Dd * 2);
    _Float16* Qh   = (_Float16*)alloc((size_t)NTt * Dd * 2);
    _Float16* W1Th = (_Float16*)alloc((size_t)Hh  * Dd * 2);
    _Float16* HXh  = (_Float16*)alloc((size_t)Nn  * Hh * 2);
    _Float16* HZh  = (_Float16*)alloc((size_t)Mm  * Hh * 2);
    _Float16* HTh  = (_Float16*)alloc((size_t)NTt * Hh * 2);
    _Float16* TXh  = (_Float16*)alloc((size_t)Nn  * Hh * 2);
    _Float16* TZh  = (_Float16*)alloc((size_t)Mm  * Hh * 2);
    _Float16* TTh  = (_Float16*)alloc((size_t)NTt * Hh * 2);
    _Float16* UXh  = (_Float16*)alloc((size_t)Nn  * Hh * 2);
    _Float16* UZh  = (_Float16*)alloc((size_t)Mm  * Hh * 2);
    _Float16* UTh  = (_Float16*)alloc((size_t)NTt * Hh * 2);
    _Float16* Kzxh = (_Float16*)alloc((size_t)Mm  * Nn * 2);

    float* Apre   = (float*)alloc((size_t)Nn * Hh * 4);   // reused for X/Z/Xtest
    float* lam    = (float*)alloc((size_t)Nn * Cc * 4);
    float* ZX     = (float*)alloc((size_t)Mm * Nn * 4);
    float* HZX    = (float*)alloc((size_t)Mm * Nn * 4);
    float* GZX    = (float*)alloc((size_t)Mm * Nn * 4);
    float* ZZ     = (float*)alloc((size_t)Mm * Mm * 4);
    float* HZZ    = (float*)alloc((size_t)Mm * Mm * 4);
    float* GZZ    = (float*)alloc((size_t)Mm * Mm * 4);
    float* TZm    = (float*)alloc((size_t)NTt * Mm * 4);
    float* HTZ    = (float*)alloc((size_t)NTt * Mm * 4);
    float* GTZ    = (float*)alloc((size_t)NTt * Mm * 4);
    float* tdot   = (float*)alloc((size_t)NTt * 4);
    float* htdot  = (float*)alloc((size_t)NTt * 4);
    float* gtd    = (float*)alloc((size_t)NTt * 4);
    float* Kzx    = (float*)alloc((size_t)Mm * Nn * 4);
    float* Kzz    = (float*)alloc((size_t)Mm * Mm * 4);
    float* Ktz    = (float*)alloc((size_t)NTt * Mm * 4);
    float* Kttd   = (float*)alloc((size_t)NTt * 4);
    float* betaR  = (float*)alloc((size_t)Mm * Mm * 4);
    float* Bmat   = (float*)alloc((size_t)Mm * Mm * 4);
    float* Binv   = (float*)alloc((size_t)Mm * Mm * 4);
    float* Kzzinv = (float*)alloc((size_t)Mm * Mm * 4);
    float* tmpm   = (float*)alloc((size_t)Mm * Mm * 4);
    float* alphaV = (float*)alloc((size_t)Mm * 4);
    float* vvec   = (float*)alloc((size_t)Mm * 4);

    // --- f16 casts + W1 transpose ---
    hipLaunchKernelGGL(cast_f32_f16, gsz(Nn * Dd),  dim3(256), 0, stream, X,  Xh, Nn * Dd);
    hipLaunchKernelGGL(cast_f32_f16, gsz(Mm * Dd),  dim3(256), 0, stream, Z,  Zh, Mm * Dd);
    hipLaunchKernelGGL(cast_f32_f16, gsz(NTt * Dd), dim3(256), 0, stream, Xt, Qh, NTt * Dd);
    hipLaunchKernelGGL(transpose_f16, gsz(Dd * Hh), dim3(256), 0, stream, W1, W1Th, Dd, Hh);

    // --- hidden activations (pre-act GEMM via WMMA, tanh epilogue) ---
    launch_gemm(Xh, W1Th, Apre, Nn, Hh, Dd, Dd, Dd, Hh, stream);
    hipLaunchKernelGGL(hidden_act, gsz(Nn * Hh), dim3(256), 0, stream, Apre, b1, HXh, TXh, Nn * Hh, Hh);
    launch_gemm(Zh, W1Th, Apre, Mm, Hh, Dd, Dd, Dd, Hh, stream);
    hipLaunchKernelGGL(hidden_act, gsz(Mm * Hh), dim3(256), 0, stream, Apre, b1, HZh, TZh, Mm * Hh, Hh);
    launch_gemm(Qh, W1Th, Apre, NTt, Hh, Dd, Dd, Dd, Hh, stream);
    hipLaunchKernelGGL(hidden_act, gsz(NTt * Hh), dim3(256), 0, stream, Apre, b1, HTh, TTh, NTt * Hh, Hh);

    // --- lambda_1 ---
    hipLaunchKernelGGL(lambda_kernel, gsz(Nn * Cc), dim3(256), 0, stream, HXh, W2, b2, Y, lam, Nn, Hh, Cc);

    // --- class-independent dot-product matrices ---
    launch_gemm(Zh,  Xh,  ZX,  Mm,  Nn, Dd, Dd, Dd, Nn, stream);
    launch_gemm(HZh, HXh, HZX, Mm,  Nn, Hh, Hh, Hh, Nn, stream);
    launch_gemm(Zh,  Zh,  ZZ,  Mm,  Mm, Dd, Dd, Dd, Mm, stream);
    launch_gemm(HZh, HZh, HZZ, Mm,  Mm, Hh, Hh, Hh, Mm, stream);
    launch_gemm(Qh,  Zh,  TZm, NTt, Mm, Dd, Dd, Dd, Mm, stream);
    launch_gemm(HTh, HZh, HTZ, NTt, Mm, Hh, Hh, Hh, Mm, stream);
    hipLaunchKernelGGL(rowsq_f32, gsz(NTt), dim3(256), 0, stream, Xt, tdot, NTt, Dd);
    hipLaunchKernelGGL(rowsq_f16, gsz(NTt), dim3(256), 0, stream, HTh, htdot, NTt, Hh);

    // --- per-class pipeline ---
    for (int cls = 0; cls < Cc; ++cls) {
        hipLaunchKernelGGL(ubuild, gsz(Nn * Hh),  dim3(256), 0, stream, TXh, W2, cls, Cc, UXh, Nn * Hh,  Hh);
        hipLaunchKernelGGL(ubuild, gsz(Mm * Hh),  dim3(256), 0, stream, TZh, W2, cls, Cc, UZh, Mm * Hh,  Hh);
        hipLaunchKernelGGL(ubuild, gsz(NTt * Hh), dim3(256), 0, stream, TTh, W2, cls, Cc, UTh, NTt * Hh, Hh);

        launch_gemm(UZh, UXh, GZX, Mm,  Nn, Hh, Hh, Hh, Nn, stream);
        launch_gemm(UZh, UZh, GZZ, Mm,  Mm, Hh, Hh, Hh, Mm, stream);
        launch_gemm(UTh, UZh, GTZ, NTt, Mm, Hh, Hh, Hh, Mm, stream);

        hipLaunchKernelGGL(assemble_k, gsz(Mm * Nn),  dim3(256), 0, stream, ZX,  HZX, GZX, Kzx, Mm * Nn,  Nn, SCALEF, 0.0f,    0);
        hipLaunchKernelGGL(assemble_k, gsz(Mm * Mm),  dim3(256), 0, stream, ZZ,  HZZ, GZZ, Kzz, Mm * Mm,  Mm, SCALEF, JITTERF, 1);
        hipLaunchKernelGGL(assemble_k, gsz(NTt * Mm), dim3(256), 0, stream, TZm, HTZ, GTZ, Ktz, NTt * Mm, Mm, SCALEF, 0.0f,    0);

        hipLaunchKernelGGL(rowsq_f16, gsz(NTt), dim3(256), 0, stream, UTh, gtd, NTt, Hh);
        hipLaunchKernelGGL(ktt_diag,  gsz(NTt), dim3(256), 0, stream, tdot, htdot, gtd, Kttd, NTt, SCALEF);

        hipLaunchKernelGGL(alpha_kernel, dim3(1), dim3(128), 0, stream, Kzx, lam, cls, Cc, alphaV, Mm, Nn);

        // beta = l2d * Kzx @ Kzx^T via WMMA (f16 operand copy of Kzx)
        hipLaunchKernelGGL(cast_f32_f16, gsz(Mm * Nn), dim3(256), 0, stream, Kzx, Kzxh, Mm * Nn);
        launch_gemm(Kzxh, Kzxh, betaR, Mm, Mm, Nn, Nn, Nn, Mm, stream);
        hipLaunchKernelGGL(combine_B, gsz(Mm * Mm), dim3(256), 0, stream, Kzz, betaR, L2D, Bmat, Mm * Mm);

        hipLaunchKernelGGL(chol_inverse_128, dim3(1), dim3(128), 0, stream, Bmat, Binv);
        hipLaunchKernelGGL(chol_inverse_128, dim3(1), dim3(128), 0, stream, Kzz,  Kzzinv);

        hipLaunchKernelGGL(sub_mat,   gsz(Mm * Mm), dim3(256), 0, stream, Kzzinv, Binv, tmpm, Mm * Mm);
        hipLaunchKernelGGL(matvec128, dim3(1), dim3(128), 0, stream, Binv, alphaV, vvec);

        hipLaunchKernelGGL(mean_var, dim3(1), dim3(256), 0, stream, Ktz, vvec, tmpm, Kttd, out, cls);
    }
}
